// Attention_66563403153646
// MI455X (gfx1250) — matrix operand order repaired and checked by reading, not compiled, not run on hardware
//
#include <hip/hip_runtime.h>
#include <cstdint>

typedef __bf16 bf16_t;
typedef __attribute__((ext_vector_type(16))) __bf16 v16bf;
typedef __attribute__((ext_vector_type(8)))  float  v8f;

constexpr int kB = 2, kS = 2048, kD = 4096, kH = 32, kHD = 128, kAL = 10;
constexpr float kInvSqrtHD = 0.08838834764831845f; // 1/sqrt(128)

union Frag16 { v16bf v; uint4 u[2]; };

__device__ __forceinline__ bf16_t f2bf(float f) { return (bf16_t)f; }
__device__ __forceinline__ float  bf2f(bf16_t b) { return (float)b; }

static __device__ __forceinline__ v8f wmma_bf16(v16bf a, v16bf b, v8f c) {
  // D = A(16x32 bf16) x B(32x16 bf16) + C(16x16 f32)
  return __builtin_amdgcn_wmma_f32_16x16x32_bf16(false, a, false, b, (short)0, c,
                                                 false, false);
}

// ---------------- stage 0: conversions ----------------
__global__ void k_cvt_bf16(const float* __restrict__ in, bf16_t* __restrict__ out, long n) {
  long i = (long)blockIdx.x * blockDim.x + threadIdx.x;
  if (i < n) out[i] = f2bf(in[i]);
}

// W (K x N row-major, fp32) -> WT (N x K row-major, bf16), 32x32 LDS tiles
__global__ void k_transpose_cvt(const float* __restrict__ W, bf16_t* __restrict__ WT) {
  __shared__ float t[32][33];
  const int tx = threadIdx.x, ty = threadIdx.y;
  const int n0 = blockIdx.x * 32, k0 = blockIdx.y * 32;
  for (int yy = 0; yy < 4; ++yy)
    t[ty + yy * 8][tx] = W[(long)(k0 + ty + yy * 8) * kD + n0 + tx];
  __syncthreads();
  for (int yy = 0; yy < 4; ++yy)
    WT[(long)(n0 + ty + yy * 8) * kD + k0 + tx] = f2bf(t[tx][ty + yy * 8]);
}

// ---------------- bf16 WMMA GEMM: C = A(MxK) * BT(NxK)^T ----------------
template <bool OUTF32>
__global__ __launch_bounds__(256) void k_gemm(const bf16_t* __restrict__ A,
                                              const bf16_t* __restrict__ BT,
                                              void* __restrict__ Cv,
                                              int M, int N, int K) {
  __shared__ bf16_t at[128][32];
  __shared__ bf16_t bt[128][32];
  const int lane = threadIdx.x & 31, wave = threadIdx.x >> 5;
  const int lo = lane & 15, hi = lane >> 4;
  const int wm = wave >> 1, wn = wave & 1;           // 4x2 wave grid
  const long rowA0 = (long)blockIdx.y * 128;
  const long rowB0 = (long)blockIdx.x * 128;
  const int lr = threadIdx.x >> 1, lseg = (threadIdx.x & 1) * 16;
  v8f acc[2][4] = {};
  for (int k0 = 0; k0 < K; k0 += 32) {
    __syncthreads();
    {
      const uint4* ga = (const uint4*)(A + (rowA0 + lr) * K + k0 + lseg);
      *(uint4*)&at[lr][lseg]     = ga[0];
      *(uint4*)&at[lr][lseg + 8] = ga[1];
      const uint4* gb = (const uint4*)(BT + (rowB0 + lr) * K + k0 + lseg);
      *(uint4*)&bt[lr][lseg]     = gb[0];
      *(uint4*)&bt[lr][lseg + 8] = gb[1];
    }
    __syncthreads();
    Frag16 af[2], bfr[4];
    for (int i = 0; i < 2; ++i) {
      const int m = wm * 32 + i * 16 + lo;
      af[i].u[0] = *(const uint4*)&at[m][hi * 8];        // K = hi*8..+7
      af[i].u[1] = *(const uint4*)&at[m][16 + hi * 8];   // K = 16+hi*8..+7
    }
    for (int j = 0; j < 4; ++j) {
      const int n = wn * 64 + j * 16 + lo;
      bfr[j].u[0] = *(const uint4*)&bt[n][hi * 8];       // K = hi*8..+7
      bfr[j].u[1] = *(const uint4*)&bt[n][16 + hi * 8];
    }
    for (int i = 0; i < 2; ++i)
      for (int j = 0; j < 4; ++j)
        acc[i][j] = wmma_bf16(af[i].v, bfr[j].v, acc[i][j]);
  }
  for (int i = 0; i < 2; ++i)
    for (int j = 0; j < 4; ++j)
      for (int r = 0; r < 8; ++r) {
        const long m = rowA0 + wm * 32 + i * 16 + r + 8 * hi;
        const long n = rowB0 + wn * 64 + j * 16 + lo;
        const float v = acc[i][j][r];
        if (OUTF32) ((float*)Cv)[m * (long)N + n] = v;
        else        ((bf16_t*)Cv)[m * (long)N + n] = f2bf(v);
      }
}

// ---------------- stage 2: RoPE (q scaled by 1/sqrt(HD)) + (b,h,s,d) layout ----
__global__ void k_rope(const bf16_t* __restrict__ qin, const bf16_t* __restrict__ kin,
                       const float* __restrict__ cosp, const float* __restrict__ sinp,
                       bf16_t* __restrict__ qout, bf16_t* __restrict__ kout) {
  const long tid = (long)blockIdx.x * blockDim.x + threadIdx.x; // B*S*H*64
  const int i = (int)(tid & 63);
  const int h = (int)((tid >> 6) & (kH - 1));
  const int s = (int)((tid >> 11) & (kS - 1));
  const int b = (int)(tid >> 22);
  const float c  = cosp[s * 64 + i];
  const float sn = sinp[s * 64 + i];
  const long inIdx  = ((long)(b * kS + s)) * kD + h * kHD + 2 * i;
  const long outIdx = (((long)(b * kH + h)) * kS + s) * kHD + 2 * i;
  const float q0 = bf2f(qin[inIdx]), q1 = bf2f(qin[inIdx + 1]);
  const float ka = bf2f(kin[inIdx]), kb = bf2f(kin[inIdx + 1]);
  qout[outIdx]     = f2bf((q0 * c - q1 * sn) * kInvSqrtHD);
  qout[outIdx + 1] = f2bf((q0 * sn + q1 * c) * kInvSqrtHD);
  kout[outIdx]     = f2bf(ka * c - kb * sn);
  kout[outIdx + 1] = f2bf(ka * sn + kb * c);
}

// V (b,s,h,d) -> VT (b,h,d,s)
__global__ void k_vtrans(const bf16_t* __restrict__ vin, bf16_t* __restrict__ vout) {
  __shared__ bf16_t t[32][33];
  const int tx = threadIdx.x, ty = threadIdx.y;
  const int d0 = blockIdx.x * 32, s0 = blockIdx.y * 32, bh = blockIdx.z;
  const int b = bh / kH, h = bh % kH;
  for (int yy = 0; yy < 4; ++yy) {
    const int s = s0 + ty + yy * 8;
    t[ty + yy * 8][tx] = vin[((long)(b * kS + s)) * kD + h * kHD + d0 + tx];
  }
  __syncthreads();
  for (int yy = 0; yy < 4; ++yy) {
    const int d = d0 + ty + yy * 8;
    vout[((long)bh * kHD + d) * kS + s0 + tx] = t[tx][ty + yy * 8];
  }
}

// ---------------- adapter K/V projection (tiny, fp32) ----------------
__global__ void k_adapter_kv(const float* __restrict__ adapter, const float* __restrict__ wk,
                             const float* __restrict__ wv, float* __restrict__ akv) {
  const int tid = blockIdx.x * blockDim.x + threadIdx.x;  // 2*AL*D
  const int which = tid / (kAL * kD);
  const int rem = tid % (kAL * kD);
  const int l = rem / kD, n = rem % kD;
  const float* w = which ? wv : wk;
  const float* a = adapter + (long)l * kD;
  float s = 0.f;
  for (int k = 0; k < kD; ++k) s += a[k] * w[(long)k * kD + n];
  akv[tid] = s;
}

// ---------------- stage 3: causal flash attention (bf16 WMMA) ----------------
__global__ __launch_bounds__(256) void k_flash(const bf16_t* __restrict__ qbf,
                                               const bf16_t* __restrict__ kbf,
                                               const bf16_t* __restrict__ vtbf,
                                               bf16_t* __restrict__ obf) {
  __shared__ bf16_t ktile[32][128];      // [key][dim]
  __shared__ bf16_t vtile[128][32];      // [dim][key]
  __shared__ bf16_t ptile[8][16][32];    // per-wave P staging (C-layout -> A-layout)
  const int bh = blockIdx.y;
  const int b = bh / kH, h = bh % kH;
  const int wave = threadIdx.x >> 5, lane = threadIdx.x & 31;
  const int lo = lane & 15, hi = lane >> 4;
  const int qrow0 = blockIdx.x * 128 + wave * 16;

  Frag16 qa[4];
  {
    const bf16_t* qp = qbf + ((long)bh * kS + qrow0 + lo) * kHD;
    for (int c = 0; c < 4; ++c) {
      qa[c].u[0] = *(const uint4*)(qp + c * 32 + hi * 8);
      qa[c].u[1] = *(const uint4*)(qp + c * 32 + 16 + hi * 8);
    }
  }
  v8f acc[8] = {};
  float rm[8], rl[8];
  for (int r = 0; r < 8; ++r) { rm[r] = -1e30f; rl[r] = 0.f; }

  const int nkb = blockIdx.x * 4 + 4;       // causal: keys up to block's last row
  for (int kblk = 0; kblk < nkb; ++kblk) {
    const int k0 = kblk * 32;
    __syncthreads();
    {
      const int key = threadIdx.x >> 3, seg = (threadIdx.x & 7) * 16;
      const uint4* g = (const uint4*)(kbf + ((long)bh * kS + k0 + key) * kHD + seg);
      *(uint4*)&ktile[key][seg]     = g[0];
      *(uint4*)&ktile[key][seg + 8] = g[1];
      const int dim = threadIdx.x >> 1, vseg = (threadIdx.x & 1) * 16;
      const uint4* gv = (const uint4*)(vtbf + ((long)bh * kHD + dim) * kS + k0 + vseg);
      *(uint4*)&vtile[dim][vseg]     = gv[0];
      *(uint4*)&vtile[dim][vseg + 8] = gv[1];
    }
    __syncthreads();

    v8f s0 = {}, s1 = {};
    for (int c = 0; c < 4; ++c) {
      Frag16 kb0, kb1;
      kb0.u[0] = *(const uint4*)&ktile[lo][c * 32 + hi * 8];
      kb0.u[1] = *(const uint4*)&ktile[lo][c * 32 + 16 + hi * 8];
      kb1.u[0] = *(const uint4*)&ktile[lo + 16][c * 32 + hi * 8];
      kb1.u[1] = *(const uint4*)&ktile[lo + 16][c * 32 + 16 + hi * 8];
      s0 = wmma_bf16(qa[c].v, kb0.v, s0);
      s1 = wmma_bf16(qa[c].v, kb1.v, s1);
    }

    // online softmax in C-layout: element (r,lane) -> M = r + 8*hi, N = lo
    for (int r = 0; r < 8; ++r) {
      const int mg = qrow0 + r + 8 * hi;
      const int key0 = k0 + lo, key1 = k0 + 16 + lo;
      float v0 = (key0 <= mg) ? s0[r] : -1e30f;
      float v1 = (key1 <= mg) ? s1[r] : -1e30f;
      float mx = fmaxf(v0, v1);
      mx = fmaxf(mx, __shfl_xor(mx, 1));
      mx = fmaxf(mx, __shfl_xor(mx, 2));
      mx = fmaxf(mx, __shfl_xor(mx, 4));
      mx = fmaxf(mx, __shfl_xor(mx, 8));
      const float mo = rm[r];
      const float mn = fmaxf(mo, mx);
      rm[r] = mn;
      const float alpha = __expf(mo - mn);
      float p0 = (key0 <= mg) ? __expf(v0 - mn) : 0.f;
      float p1 = (key1 <= mg) ? __expf(v1 - mn) : 0.f;
      float rs = p0 + p1;
      rs += __shfl_xor(rs, 1);
      rs += __shfl_xor(rs, 2);
      rs += __shfl_xor(rs, 4);
      rs += __shfl_xor(rs, 8);
      rl[r] = rl[r] * alpha + rs;
      for (int d = 0; d < 8; ++d) acc[d][r] *= alpha;
      ptile[wave][r + 8 * hi][lo]      = f2bf(p0);
      ptile[wave][r + 8 * hi][16 + lo] = f2bf(p1);
    }
    asm volatile("s_wait_dscnt 0" ::: "memory");  // per-wave LDS RAW fence

    Frag16 pa;  // A-layout: lane row = lo, K = {hi*8..+7, 16+hi*8..+7}
    pa.u[0] = *(const uint4*)&ptile[wave][lo][hi * 8];
    pa.u[1] = *(const uint4*)&ptile[wave][lo][16 + hi * 8];
    for (int d = 0; d < 8; ++d) {
      Frag16 vb;
      vb.u[0] = *(const uint4*)&vtile[d * 16 + lo][hi * 8];
      vb.u[1] = *(const uint4*)&vtile[d * 16 + lo][16 + hi * 8];
      acc[d] = wmma_bf16(pa.v, vb.v, acc[d]);
    }
  }

  for (int d = 0; d < 8; ++d)
    for (int r = 0; r < 8; ++r) {
      const int mg = qrow0 + r + 8 * hi;
      const float o = acc[d][r] / rl[r];
      obf[(((long)(b * kS + mg)) * kH + h) * kHD + d * 16 + lo] = f2bf(o);
    }
}

// ---------------- stage 3b: gated adapter attention, RMW into O ----------------
__global__ void k_adapter_apply(const bf16_t* __restrict__ qbf, const float* __restrict__ akv,
                                const float* __restrict__ gate, bf16_t* __restrict__ obf) {
  const int tid = blockIdx.x * blockDim.x + threadIdx.x;  // B*H*S
  const int s = tid & (kS - 1);
  const int h = (tid >> 11) & (kH - 1);
  const int b = tid >> 16;
  const bf16_t* q = qbf + (((long)(b * kH + h)) * kS + s) * kHD;  // roped+scaled
  float sc[kAL];
  float mx = -1e30f;
  for (int l = 0; l < kAL; ++l) {
    const float* ak = akv + (long)l * kD + h * kHD;
    float t = 0.f;
    for (int d = 0; d < kHD; ++d) t += bf2f(q[d]) * ak[d];
    sc[l] = t; mx = fmaxf(mx, t);
  }
  float den = 0.f;
  for (int l = 0; l < kAL; ++l) { sc[l] = __expf(sc[l] - mx); den += sc[l]; }
  const float g = gate[h] / den;
  bf16_t* o = obf + (((long)(b * kS + s)) * kH + h) * kHD;
  const float* av = akv + (long)kAL * kD + h * kHD;
  for (int d = 0; d < kHD; ++d) {
    float a = bf2f(o[d]);
    for (int l = 0; l < kAL; ++l) a += g * sc[l] * av[(long)l * kD + d];
    o[d] = f2bf(a);
  }
}

// ---------------- host-side orchestration ----------------
extern "C" void kernel_launch(void* const* d_in, const int* in_sizes, int n_in,
                              void* d_out, int out_size, void* d_ws, size_t ws_size,
                              hipStream_t stream) {
  (void)in_sizes; (void)n_in; (void)out_size; (void)ws_size;
  const float* x       = (const float*)d_in[0];
  const float* cosp    = (const float*)d_in[1];
  const float* sinp    = (const float*)d_in[2];
  // d_in[3] mask: causal triu(-1e9) generated analytically in-kernel
  const float* wq      = (const float*)d_in[4];
  const float* wk      = (const float*)d_in[5];
  const float* wv      = (const float*)d_in[6];
  const float* wo      = (const float*)d_in[7];
  const float* gate    = (const float*)d_in[8];
  const float* adapter = (const float*)d_in[9];
  // d_in[10] random_init == 0

  char* ws = (char*)d_ws;
  const long NE = (long)kB * kS * kD;            // 16,777,216 elements
  bf16_t* xbf = (bf16_t*)ws; ws += NE * 2;
  bf16_t* wqT = (bf16_t*)ws; ws += (long)kD * kD * 2;
  bf16_t* wkT = (bf16_t*)ws; ws += (long)kD * kD * 2;
  bf16_t* wvT = (bf16_t*)ws; ws += (long)kD * kD * 2;
  bf16_t* woT = (bf16_t*)ws; ws += (long)kD * kD * 2;
  bf16_t* qkv = (bf16_t*)ws; ws += 3 * NE * 2;   // q|k|v pre-rope; reused as O
  bf16_t* qro = (bf16_t*)ws; ws += NE * 2;
  bf16_t* kro = (bf16_t*)ws; ws += NE * 2;
  bf16_t* vtr = (bf16_t*)ws; ws += NE * 2;
  float*  akv = (float*)ws;  ws += (long)2 * kAL * kD * 4;
  bf16_t* obf = qkv;  // qkv dead after rope/vtrans; alias for O

  const int M = kB * kS, N = kD, K = kD;

  // stage 0: conversions
  k_cvt_bf16<<<(int)((NE + 255) / 256), 256, 0, stream>>>(x, xbf, NE);
  dim3 tb(32, 8), tg(kD / 32, kD / 32);
  k_transpose_cvt<<<tg, tb, 0, stream>>>(wq, wqT);
  k_transpose_cvt<<<tg, tb, 0, stream>>>(wk, wkT);
  k_transpose_cvt<<<tg, tb, 0, stream>>>(wv, wvT);
  k_transpose_cvt<<<tg, tb, 0, stream>>>(wo, woT);

  // stage 1: QKV projections (bf16 WMMA)
  dim3 gg(N / 128, M / 128);
  k_gemm<false><<<gg, 256, 0, stream>>>(xbf, wqT, qkv + 0 * NE, M, N, K);
  k_gemm<false><<<gg, 256, 0, stream>>>(xbf, wkT, qkv + 1 * NE, M, N, K);
  k_gemm<false><<<gg, 256, 0, stream>>>(xbf, wvT, qkv + 2 * NE, M, N, K);

  // adapter K/V (tiny fp32)
  k_adapter_kv<<<(2 * kAL * kD) / 256, 256, 0, stream>>>(adapter, wk, wv, akv);

  // stage 2: RoPE (+1/sqrt(HD) into Q) and V transpose
  const long nrope = (long)kB * kS * kH * (kHD / 2);
  k_rope<<<(int)((nrope + 255) / 256), 256, 0, stream>>>(qkv + 0 * NE, qkv + 1 * NE,
                                                         cosp, sinp, qro, kro);
  dim3 vt_g(kHD / 32, kS / 32, kB * kH);
  k_vtrans<<<vt_g, tb, 0, stream>>>(qkv + 2 * NE, vtr);

  // stage 3: causal flash attention, then gated adapter add
  dim3 fg(kS / 128, kB * kH);
  k_flash<<<fg, 256, 0, stream>>>(qro, kro, vtr, obf);
  k_adapter_apply<<<(kB * kH * kS) / 256, 256, 0, stream>>>(qro, akv, gate, obf);

  // stage 4: output projection (fp32 epilogue into d_out)
  k_gemm<true><<<gg, 256, 0, stream>>>(obf, woT, d_out, M, N, K);
}
